// MemoryEfficientAttention_24215025615661
// MI455X (gfx1250) — matrix operand order repaired and checked by reading, not compiled, not run on hardware
//
#include <hip/hip_runtime.h>
#include <hip/hip_bf16.h>

// Problem constants (from reference): B=2, S=2048, E=1024, H=16, D=64
#define BQ   2
#define SEQ  2048
#define EMB  1024
#define NH   16
#define HD   64
#define MROWS (BQ * SEQ)   // 4096
#define QSCALE 0.125f      // HEAD_DIM^-0.5

typedef __attribute__((ext_vector_type(16))) __bf16 v16bf;
typedef __attribute__((ext_vector_type(8)))  __bf16 v8bf;
typedef __attribute__((ext_vector_type(8)))  float  v8f;
typedef __attribute__((ext_vector_type(4)))  float  v4f;
typedef __attribute__((ext_vector_type(4)))  unsigned int v4u;
typedef __attribute__((ext_vector_type(8)))  unsigned int v8u;

static __device__ __forceinline__ v8f wmma_bf16(v16bf a, v16bf b, v8f c) {
  return __builtin_amdgcn_wmma_f32_16x16x32_bf16(false, a, false, b,
                                                 (short)0, c, false, false);
}

// ---- Tensor Data Mover: 4 KB contiguous global -> LDS copy -----------------
// D# per CDNA5 ISA: 2D tensor, data_size=2B,
// tile 2048x1, tensor_dim0 = tensor_dim0_stride = 2048.  Issued once per wave
// (EXEC ignored); completion tracked with TENSORcnt (in-order per wave).
static __device__ __forceinline__ void tdm_load_tile(const __bf16* gptr,
                                                     unsigned lds_off) {
  const unsigned long long ga = (unsigned long long)(uintptr_t)gptr;
  v4u g0;
  g0[0] = 1u;                                   // count=1 (valid user D#)
  g0[1] = lds_off;                              // lds_addr (bytes)
  g0[2] = (unsigned)(ga & 0xffffffffu);         // global_addr[31:0]
  g0[3] = (unsigned)((ga >> 32) & 0x01ffffffu)  // global_addr[56:32]
        | (2u << 30);                           // type = 2 ("image")
  v8u g1;
  g1[0] = 1u << 16;                             // wg_mask=0, data_size=1 (2B)
  g1[1] = 2048u << 16;                          // tensor_dim0[15:0] @ bits63:48
  g1[2] = 0u | (1u << 16);                      // tensor_dim0 hi=0; tensor_dim1=1
  g1[3] = 0u | (2048u << 16);                   // tensor_dim1 hi=0; tile_dim0=2048
  g1[4] = 1u;                                   // tile_dim1=1, tile_dim2=0
  g1[5] = 2048u;                                // tensor_dim0_stride[31:0]
  g1[6] = 0u;                                   // stride hi, tensor_dim1_stride lo
  g1[7] = 0u;
  asm volatile("tensor_load_to_lds %0, %1" :: "s"(g0), "s"(g1) : "memory");
}

// ---- Fragment loaders (CDNA5 16x16x32 bf16 VGPR layouts) -------------------
// A (16x32, MxK): lanes 0-15 -> rows M=lane, elems 0..7 = K 0..7, 8..15 = K 16..23
//                 lanes 16-31 -> rows M=lane-16, K shifted by +8.
static __device__ __forceinline__ v16bf
load_a_f32(const float* __restrict__ X, int ld, int row0, int k0, int lane) {
  const int r  = row0 + (lane & 15);
  const int kh = (lane >> 4) * 8;
  const float* p = X + (size_t)r * ld + k0 + kh;
  union { v4f v[4]; float s[16]; } u;
  u.v[0] = *reinterpret_cast<const v4f*>(p);
  u.v[1] = *reinterpret_cast<const v4f*>(p + 4);
  u.v[2] = *reinterpret_cast<const v4f*>(p + 16);
  u.v[3] = *reinterpret_cast<const v4f*>(p + 20);
  v16bf a;
#pragma unroll
  for (int j = 0; j < 16; ++j) a[j] = (__bf16)u.s[j];
  return a;
}

static __device__ __forceinline__ v16bf
load_a_bf16(const __bf16* __restrict__ X, int ld, int row0, int k0, int lane) {
  const int r  = row0 + (lane & 15);
  const int kh = (lane >> 4) * 8;
  const __bf16* p = X + (size_t)r * ld + k0 + kh;
  v8bf lo = *reinterpret_cast<const v8bf*>(p);
  v8bf hi = *reinterpret_cast<const v8bf*>(p + 16);
  v16bf a;
#pragma unroll
  for (int j = 0; j < 8; ++j) { a[j] = lo[j]; a[8 + j] = hi[j]; }
  return a;
}

// B (32x16, KxN): lanes 0-15 -> col N=lane, elems 0..7 = K 0..7, 8..15 = K 16..23; lanes 16-31 -> K shifted by +8.
// B[k][n] = Mrow[n][k] (row-major source with contiguous K): two 16 B loads.
static __device__ __forceinline__ v16bf
load_b_bf16(const __bf16* __restrict__ X, int ld, int n0, int k0, int lane) {
  const int n  = n0 + (lane & 15);
  const int kh = (lane >> 4) * 8;
  const __bf16* p = X + (size_t)n * ld + k0 + kh;
  v8bf lo = *reinterpret_cast<const v8bf*>(p);
  v8bf hi = *reinterpret_cast<const v8bf*>(p + 16);
  v16bf b;
#pragma unroll
  for (int j = 0; j < 8; ++j) { b[j] = lo[j]; b[8 + j] = hi[j]; }
  return b;
}

// B from row-major V tile [32 keys][64 d]: B[k][n] = Vt[k][n] -> strided (LDS).
static __device__ __forceinline__ v16bf
load_b_bf16_colmajor(const __bf16* __restrict__ Vt, int ld, int n0, int lane) {
  const int n  = n0 + (lane & 15);
  const int kh = (lane >> 4) * 8;
  v16bf b;
#pragma unroll
  for (int j = 0; j < 16; ++j) b[j] = Vt[(size_t)(kh + (j < 8 ? j : j + 8)) * ld + n];
  return b;
}

// Cross-lane row reductions inside the 16-lane half-wave (C/D row groups).
static __device__ __forceinline__ float rowmax16(float x) {
  x = fmaxf(x, __shfl_xor(x, 1, 32));
  x = fmaxf(x, __shfl_xor(x, 2, 32));
  x = fmaxf(x, __shfl_xor(x, 4, 32));
  x = fmaxf(x, __shfl_xor(x, 8, 32));
  return x;
}
static __device__ __forceinline__ float rowsum16(float x) {
  x += __shfl_xor(x, 1, 32);
  x += __shfl_xor(x, 2, 32);
  x += __shfl_xor(x, 4, 32);
  x += __shfl_xor(x, 8, 32);
  return x;
}

// ---- Kernel 0: one-shot f32 -> bf16 weight conversion ----------------------
__global__ __launch_bounds__(256)
void cvt_bf16_kernel(const float* __restrict__ X, __bf16* __restrict__ Y) {
  const size_t i = ((size_t)blockIdx.x * 256 + threadIdx.x) * 8;
  v4f f0 = *reinterpret_cast<const v4f*>(X + i);
  v4f f1 = *reinterpret_cast<const v4f*>(X + i + 4);
  v8bf o;
#pragma unroll
  for (int j = 0; j < 4; ++j) { o[j] = (__bf16)f0[j]; o[4 + j] = (__bf16)f1[j]; }
  *reinterpret_cast<v8bf*>(Y + i) = o;
}

// ---- Kernel 1: QKV projection. Y[b,h,s,d] = bf16( (X @ W^T + bias) * scale )
// Wave tile: 32(M) x 64(N); block: 8 waves stacked in M -> 256 x 64.
__global__ __launch_bounds__(256)
void proj_kernel(const float* __restrict__ X, const __bf16* __restrict__ Wb,
                 const float* __restrict__ bias, __bf16* __restrict__ Y,
                 float out_scale) {
  const int lane = threadIdx.x & 31;
  const int wave = threadIdx.x >> 5;
  const int m0 = blockIdx.x * 256 + wave * 32;
  const int n0 = blockIdx.y * 64;
  v8f acc[2][4] = {};
  for (int k0 = 0; k0 < EMB; k0 += 32) {
    v16bf a0 = load_a_f32(X, EMB, m0, k0, lane);
    v16bf a1 = load_a_f32(X, EMB, m0 + 16, k0, lane);
#pragma unroll
    for (int nt = 0; nt < 4; ++nt) {
      v16bf bfr = load_b_bf16(Wb, EMB, n0 + nt * 16, k0, lane);
      acc[0][nt] = wmma_bf16(a0, bfr, acc[0][nt]);
      acc[1][nt] = wmma_bf16(a1, bfr, acc[1][nt]);
    }
  }
#pragma unroll
  for (int nt = 0; nt < 4; ++nt) {
    const int n = n0 + nt * 16 + (lane & 15);
    const int h = n >> 6, d = n & (HD - 1);
    const float bn = bias[n];
#pragma unroll
    for (int mi = 0; mi < 2; ++mi) {
#pragma unroll
      for (int r = 0; r < 8; ++r) {
        const int m = m0 + mi * 16 + r + ((lane >> 4) << 3);
        const int b = m >> 11, s = m & (SEQ - 1);
        const float v = (acc[mi][nt][r] + bn) * out_scale;
        Y[(((size_t)(b * NH + h) * SEQ + s) << 6) + d] = (__bf16)v;
      }
    }
  }
}

// ---- Kernel 2: flash attention over [B*H] x [S/128] blocks -----------------
// K/V tiles double-buffered in LDS via the Tensor Data Mover; wave 0 issues
// next-tile DMAs while all waves compute on the current tile.
__global__ __launch_bounds__(256)
void attn_kernel(const __bf16* __restrict__ Q, const __bf16* __restrict__ K,
                 const __bf16* __restrict__ V, __bf16* __restrict__ O) {
  __shared__ alignas(16) __bf16 Kt[2][32 * HD];    // 2 x 4 KB key tiles
  __shared__ alignas(16) __bf16 Vt[2][32 * HD];    // 2 x 4 KB value tiles
  __shared__ alignas(32) __bf16 Pb[8][16 * 32];    // per-wave P staging

  const int lane = threadIdx.x & 31;
  const int wave = threadIdx.x >> 5;
  const int bh   = blockIdx.y;                      // 0..31
  const int q0   = blockIdx.x * 128 + wave * 16;
  const size_t base = (size_t)bh * SEQ * HD;
  const __bf16* Kg = K + base;
  const __bf16* Vg = V + base;

  // Q rows for this wave, as two A fragments (K-dim 64 = 2 x 32)
  const v16bf qa0 = load_a_bf16(Q + base, HD, q0, 0, lane);
  const v16bf qa1 = load_a_bf16(Q + base, HD, q0, 32, lane);

  v8f o[4] = {};
  float mrow[8], lrow[8];
#pragma unroll
  for (int r = 0; r < 8; ++r) { mrow[r] = -1e30f; lrow[r] = 0.0f; }

  // Prime buffer 0 with the first K/V tiles.
  if (wave == 0) {
    tdm_load_tile(Kg, (unsigned)(uintptr_t)&Kt[0][0]);
    tdm_load_tile(Vg, (unsigned)(uintptr_t)&Vt[0][0]);
  }

  int buf = 0;
  for (int kc = 0; kc < SEQ; kc += 32) {
    if (wave == 0) {
      if (kc + 32 < SEQ) {
        // Prefetch next tile into the other buffer, then retire current tile:
        // TENSORcnt is in-order per wave, so <=2 means the first pair is done.
        tdm_load_tile(Kg + (size_t)(kc + 32) * HD,
                      (unsigned)(uintptr_t)&Kt[buf ^ 1][0]);
        tdm_load_tile(Vg + (size_t)(kc + 32) * HD,
                      (unsigned)(uintptr_t)&Vt[buf ^ 1][0]);
        __builtin_amdgcn_s_wait_tensorcnt(2);
      } else {
        __builtin_amdgcn_s_wait_tensorcnt(0);
      }
    }
    __syncthreads();   // publish current tile to all waves
    const __bf16* Ktb = &Kt[buf][0];
    const __bf16* Vtb = &Vt[buf][0];

    // Scores: S[16 q x 32 keys] = Q(16x64) . K^T  (SCALE folded into Q)
    v8f s0 = {}, s1 = {};
    s0 = wmma_bf16(qa0, load_b_bf16(Ktb, HD, 0, 0, lane), s0);
    s0 = wmma_bf16(qa1, load_b_bf16(Ktb, HD, 0, 32, lane), s0);
    s1 = wmma_bf16(qa0, load_b_bf16(Ktb, HD, 16, 0, lane), s1);
    s1 = wmma_bf16(qa1, load_b_bf16(Ktb, HD, 16, 32, lane), s1);

    // Online softmax update (per C-layout row r / r+8)
    float alpha[8];
#pragma unroll
    for (int r = 0; r < 8; ++r) {
      float mx = rowmax16(fmaxf(s0[r], s1[r]));
      const float mn = fmaxf(mrow[r], mx);
      alpha[r] = __expf(mrow[r] - mn);
      mrow[r] = mn;
      const float p0 = __expf(s0[r] - mn);
      const float p1 = __expf(s1[r] - mn);
      s0[r] = p0; s1[r] = p1;
      lrow[r] = lrow[r] * alpha[r] + rowsum16(p0 + p1);
    }

    // Re-layout P (C-layout -> A-layout) through per-wave LDS
    __bf16* pw = &Pb[wave][0];
#pragma unroll
    for (int r = 0; r < 8; ++r) {
      const int m = r + ((lane >> 4) << 3);
      pw[m * 32 + (lane & 15)]      = (__bf16)s0[r];
      pw[m * 32 + 16 + (lane & 15)] = (__bf16)s1[r];
    }
    const v16bf pa = load_a_bf16(pw, 32, 0, 0, lane);

    // O = O*alpha + P @ V_chunk   (K-dim 32, 4 d-tiles)
#pragma unroll
    for (int dt = 0; dt < 4; ++dt) {
#pragma unroll
      for (int r = 0; r < 8; ++r) o[dt][r] *= alpha[r];
      const v16bf vb = load_b_bf16_colmajor(Vtb, HD, dt * 16, lane);
      o[dt] = wmma_bf16(pa, vb, o[dt]);
    }
    __syncthreads();   // all reads of this buffer done before TDM reuses it
    buf ^= 1;
  }

  // Epilogue: normalize and store to [B, S, H*D] bf16
  const int b = bh >> 4, h = bh & (NH - 1);
#pragma unroll
  for (int dt = 0; dt < 4; ++dt) {
    const int d = dt * 16 + (lane & 15);
#pragma unroll
    for (int r = 0; r < 8; ++r) {
      const int qrow = q0 + r + ((lane >> 4) << 3);
      const float v = o[dt][r] / lrow[r];
      O[((size_t)b * SEQ + qrow) * EMB + h * HD + d] = (__bf16)v;
    }
  }
}

// ---- Kernel 3: output projection. out = O @ Wo^T + bo (f32) ----------------
__global__ __launch_bounds__(256)
void oproj_kernel(const __bf16* __restrict__ X, const __bf16* __restrict__ Wb,
                  const float* __restrict__ bias, float* __restrict__ Y) {
  const int lane = threadIdx.x & 31;
  const int wave = threadIdx.x >> 5;
  const int m0 = blockIdx.x * 256 + wave * 32;
  const int n0 = blockIdx.y * 64;
  v8f acc[2][4] = {};
  for (int k0 = 0; k0 < EMB; k0 += 32) {
    v16bf a0 = load_a_bf16(X, EMB, m0, k0, lane);
    v16bf a1 = load_a_bf16(X, EMB, m0 + 16, k0, lane);
#pragma unroll
    for (int nt = 0; nt < 4; ++nt) {
      v16bf bfr = load_b_bf16(Wb, EMB, n0 + nt * 16, k0, lane);
      acc[0][nt] = wmma_bf16(a0, bfr, acc[0][nt]);
      acc[1][nt] = wmma_bf16(a1, bfr, acc[1][nt]);
    }
  }
#pragma unroll
  for (int nt = 0; nt < 4; ++nt) {
    const int n = n0 + nt * 16 + (lane & 15);
    const float bn = bias[n];
#pragma unroll
    for (int mi = 0; mi < 2; ++mi) {
#pragma unroll
      for (int r = 0; r < 8; ++r) {
        const int m = m0 + mi * 16 + r + ((lane >> 4) << 3);
        Y[(size_t)m * EMB + n] = acc[mi][nt][r] + bn;
      }
    }
  }
}

extern "C" void kernel_launch(void* const* d_in, const int* in_sizes, int n_in,
                              void* d_out, int out_size, void* d_ws, size_t ws_size,
                              hipStream_t stream) {
  (void)in_sizes; (void)n_in; (void)out_size; (void)ws_size;
  const float* q  = (const float*)d_in[0];
  const float* k  = (const float*)d_in[1];
  const float* v  = (const float*)d_in[2];
  const float* Wq = (const float*)d_in[3];
  const float* bq = (const float*)d_in[4];
  const float* Wk = (const float*)d_in[5];
  const float* bk = (const float*)d_in[6];
  const float* Wv = (const float*)d_in[7];
  const float* bv = (const float*)d_in[8];
  const float* Wo = (const float*)d_in[9];
  const float* bo = (const float*)d_in[10];

  // Workspace: 4 bf16 activation tensors (32 MB) + 4 bf16 weight copies (8 MB)
  const size_t per  = (size_t)MROWS * EMB;          // 4 M elements
  const size_t wper = (size_t)EMB * EMB;            // 1 M elements
  __bf16* Qws = (__bf16*)d_ws;
  __bf16* Kws = Qws + per;
  __bf16* Vws = Kws + per;
  __bf16* Ows = Vws + per;
  __bf16* Wqb = Ows + per;
  __bf16* Wkb = Wqb + wper;
  __bf16* Wvb = Wkb + wper;
  __bf16* Wob = Wvb + wper;

  // One-shot weight conversion: 1 M elems / 8 per thread = 512 blocks of 256
  const int cvt_blocks = (int)(wper / (256 * 8));
  cvt_bf16_kernel<<<cvt_blocks, 256, 0, stream>>>(Wq, Wqb);
  cvt_bf16_kernel<<<cvt_blocks, 256, 0, stream>>>(Wk, Wkb);
  cvt_bf16_kernel<<<cvt_blocks, 256, 0, stream>>>(Wv, Wvb);
  cvt_bf16_kernel<<<cvt_blocks, 256, 0, stream>>>(Wo, Wob);

  dim3 gemm_grid(MROWS / 256, EMB / 64);   // 16 x 16 blocks, 8 waves each
  proj_kernel<<<gemm_grid, 256, 0, stream>>>(q, Wqb, bq, Qws, QSCALE);
  proj_kernel<<<gemm_grid, 256, 0, stream>>>(k, Wkb, bk, Kws, 1.0f);
  proj_kernel<<<gemm_grid, 256, 0, stream>>>(v, Wvb, bv, Vws, 1.0f);

  attn_kernel<<<dim3(SEQ / 128, BQ * NH), 256, 0, stream>>>(Qws, Kws, Vws, Ows);

  oproj_kernel<<<gemm_grid, 256, 0, stream>>>(Ows, Wob, bo, (float*)d_out);
}
